// MHMS_17119739642208
// MI455X (gfx1250) — matrix operand order repaired and checked by reading, not compiled, not run on hardware
//
#include <hip/hip_runtime.h>
#include <hip/hip_bf16.h>
#include <math.h>

typedef __attribute__((ext_vector_type(16))) _Float16 v16h;
typedef __attribute__((ext_vector_type(8)))  float    v8f;

// ---------------------------------------------------------------------------
// WMMA fragment helpers (wave32; C/D layout per CDNA5 ISA §7.12.2, A/B use contiguous K order)
// ---------------------------------------------------------------------------

__device__ __forceinline__ v8f wmma_f16(v16h a, v16h b, v8f c) {
  // 8 args: (neg_a, A, neg_b, B, c_mod, C, reuse_a, reuse_b)
  return __builtin_amdgcn_wmma_f32_16x16x32_f16(false, a, false, b, (short)0, c,
                                                false, false);
}

// A fragment: 16x32 f16, row-major source, row m = lane&15.
// VGPR v (=i/2): K = 16*(lane>=16) + 2*v + (i&1)
__device__ __forceinline__ v16h load_a_frag(const _Float16* A, int lda) {
  int lane = threadIdx.x & 31;
  int m  = lane & 15;
  int kh = (lane >> 4) << 4; // 0 or 16
  const _Float16* row = A + (size_t)m * lda;
  v16h a;
#pragma unroll
  for (int i = 0; i < 16; i += 2) {
    int v = i >> 1;
    int k = kh + (v << 1);
    a[i]     = row[k];
    a[i + 1] = row[k + 1];
  }
  return a;
}

// B fragment: 32x16 f16 where B[k][n] = W[n][k] (W is NxK row-major weights).
// lane n = lane&15, K base = 16*(lane>=16), 16 contiguous K per lane.
__device__ __forceinline__ v16h load_b_frag(const _Float16* Wtile, int ldw) {
  int lane = threadIdx.x & 31;
  int n  = lane & 15;
  int kb = (lane >> 4) << 4; // 0 or 16
  const _Float16* p = Wtile + (size_t)n * ldw + kb;
  v16h b;
#pragma unroll
  for (int i = 0; i < 16; ++i) b[i] = p[i];
  return b;
}

// C/D store: VGPR r -> D[(r + 8*(lane>=16))*ldd + (lane&15) + colbase]
__device__ __forceinline__ void store_c_bias(float* D, int ldd, int colbase,
                                             float bn, v8f c) {
  int lane = threadIdx.x & 31;
  int n  = (lane & 15) + colbase;
  int mo = (lane >> 4) << 3;
#pragma unroll
  for (int r = 0; r < 8; ++r) D[(size_t)(r + mo) * ldd + n] = c[r] + bn;
}

// ---------------------------------------------------------------------------
// Small utility kernels
// ---------------------------------------------------------------------------

__global__ __launch_bounds__(256) void k_f32_to_f16(const float* src,
                                                    _Float16* dst, long n) {
  long i = (long)blockIdx.x * blockDim.x + threadIdx.x;
  if (i < n) dst[i] = (_Float16)src[i];
}

__global__ __launch_bounds__(256) void k_zero_f32(float* p, long n) {
  long i = (long)blockIdx.x * blockDim.x + threadIdx.x;
  if (i < n) p[i] = 0.0f;
}

// Embedding gather -> f16 matrix X (65536 x 128)
__global__ __launch_bounds__(256) void k_gather(const int* ids, const float* emb,
                                                _Float16* X, long total) {
  long i = (long)blockIdx.x * blockDim.x + threadIdx.x;
  if (i >= total) return;
  long row = i >> 7;
  int  j   = (int)(i & 127);
  int  id  = ids[row];
  X[i] = (_Float16)emb[(size_t)id * 128 + j];
}

// dst[r][c] = (f16)(src[r][c] * scale[r])
__global__ __launch_bounds__(256) void k_scale_rows(const float* src,
                                                    const float* scale,
                                                    _Float16* dst, long rows,
                                                    int cols) {
  long i = (long)blockIdx.x * blockDim.x + threadIdx.x;
  if (i >= rows * (long)cols) return;
  long r = i / cols;
  dst[i] = (_Float16)(src[i] * scale[r]);
}

// Per-row L2 normalize (f32 -> f16). One wave per row.
__global__ __launch_bounds__(256) void k_rownorm(const float* src, _Float16* dst,
                                                 int rows, int cols) {
  int wave = (blockIdx.x * blockDim.x + threadIdx.x) >> 5;
  int lane = threadIdx.x & 31;
  if (wave >= rows) return;
  const float* r = src + (size_t)wave * cols;
  float s = 0.f;
  for (int j = lane; j < cols; j += 32) { float v = r[j]; s += v * v; }
#pragma unroll
  for (int off = 16; off; off >>= 1) s += __shfl_xor(s, off, 32);
  float inv = 1.f / (sqrtf(s) + 1e-12f);
  _Float16* d = dst + (size_t)wave * cols;
  for (int j = lane; j < cols; j += 32) d[j] = (_Float16)(r[j] * inv);
}

// MLP scorer tail: p = sigmoid(sum_h act(hid[r][h])*w2[h] + b2). Wave per row.
__global__ __launch_bounds__(256) void k_scorer_reduce(const float* hid,
                                                       const float* w2,
                                                       const float* b2,
                                                       float* out_a, float* out_b,
                                                       int rows, int Hc, int act) {
  int wave = (blockIdx.x * blockDim.x + threadIdx.x) >> 5;
  int lane = threadIdx.x & 31;
  if (wave >= rows) return;
  const float* hr = hid + (size_t)wave * Hc;
  float s = 0.f;
  for (int j = lane; j < Hc; j += 32) {
    float hv = hr[j];
    hv = (act == 0) ? fmaxf(hv, 0.f) : tanhf(hv);
    s += hv * w2[j];
  }
#pragma unroll
  for (int off = 16; off; off >>= 1) s += __shfl_xor(s, off, 32);
  if (lane == 0) {
    float p = 1.f / (1.f + __expf(-(s + b2[0])));
    out_a[wave] = p;
    if (out_b) out_b[wave] = p;
  }
}

// ---------------------------------------------------------------------------
// Generic (batched) WMMA GEMM: D = A (MxK f16, row-major) * W^T + bias.
// W is NxK row-major f16 (so B[k][n] = W[n][k]).
// Register-blocked: one wave computes a 32x64 output tile (2 A-frags x 4
// B-frags -> 8 WMMAs per 32-wide K chunk), amortizing operand loads ~2.7x
// vs one-tile-per-wave. Prefetch pulls K+4 chunks ahead into cache.
// Requires M % 32 == 0, N % 64 == 0, K % 32 == 0.
// ---------------------------------------------------------------------------
__global__ __launch_bounds__(256) void k_wmma_gemm(const _Float16* A,
                                                   const _Float16* W,
                                                   const float* bias, float* D,
                                                   int M, int N, int K,
                                                   size_t sA, size_t sW,
                                                   size_t sD) {
  int batch = blockIdx.y;
  A += (size_t)batch * sA;
  W += (size_t)batch * sW;
  D += (size_t)batch * sD;
  int ntg = N >> 6; // 64-col groups
  int tile = blockIdx.x * (blockDim.x >> 5) + (threadIdx.x >> 5);
  int total = (M >> 5) * ntg;
  if (tile >= total) return;
  int mt = tile / ntg; // 32-row block index
  int ng = tile % ntg; // 64-col group index
  const _Float16* A0 = A + (size_t)(mt * 32) * K;
  const _Float16* A1 = A0 + (size_t)16 * K;
  const _Float16* W0 = W + (size_t)(ng * 64) * K;

  v8f acc[2][4] = {};
  for (int k0 = 0; k0 < K; k0 += 32) {
    if (k0 + 128 < K) { // prefetch 4 chunks ahead (global_prefetch_b8)
      __builtin_prefetch(A0 + k0 + 128, 0, 0);
      __builtin_prefetch(W0 + k0 + 128, 0, 0);
    }
    v16h a0 = load_a_frag(A0 + k0, K);
    v16h a1 = load_a_frag(A1 + k0, K);
#pragma unroll
    for (int j = 0; j < 4; ++j) {
      v16h b = load_b_frag(W0 + (size_t)(j * 16) * K + k0, K);
      acc[0][j] = wmma_f16(a0, b, acc[0][j]);
      acc[1][j] = wmma_f16(a1, b, acc[1][j]);
    }
  }
  int lane = threadIdx.x & 31;
#pragma unroll
  for (int i = 0; i < 2; ++i) {
#pragma unroll
    for (int j = 0; j < 4; ++j) {
      int col = ng * 64 + j * 16;
      float bn = bias ? bias[col + (lane & 15)] : 0.f;
      store_c_bias(D + (size_t)(mt * 32 + i * 16) * N, N, col, bn, acc[i][j]);
    }
  }
}

// ---------------------------------------------------------------------------
// Persistent bidirectional GRU scan.
// Grid = 128 blocks (dir = blockIdx.x>>6), 256 threads = 8 waves.
// Each block owns 32 sequence rows; per step t, waves tile the 32x384 gate
// matrices for BOTH x_t*Wi^T and h*Wh^T via WMMA (wave -> 16 rows x 96 cols,
// A-fragments reused across 6 gate-column tiles), then a VALU phase applies
// the GRU update. h kept in LDS (f32 + f16 copy); 120KB LDS/block.
// Writes mean hidden state (text_features) f32 + f16 copy.
// ---------------------------------------------------------------------------
__global__ __launch_bounds__(256) void k_gru_scan(
    const _Float16* X,                       // (2048*32, 128) f16
    const _Float16* WiF, const _Float16* WhF,
    const float* biF, const float* bhF,
    const _Float16* WiB, const _Float16* WhB,
    const float* biB, const float* bhB,
    float* tf, _Float16* tf16) {             // (2048, 256)
  __shared__ float    h32[32 * 128];
  __shared__ _Float16 h16[32 * 128];
  __shared__ float    hg[32 * 384];
  __shared__ float    xg[32 * 384];

  const int dir     = blockIdx.x >> 6;           // 0 fwd, 1 bwd
  const int rowbase = (blockIdx.x & 63) * 32;    // sequence rows
  const _Float16* Wi = dir ? WiB : WiF;
  const _Float16* Wh = dir ? WhB : WhF;
  const float*    bi = dir ? biB : biF;
  const float*    bh = dir ? bhB : bhF;

  const int tid  = threadIdx.x;
  const int wave = tid >> 5;
  const int lane = tid & 31;
  const int wrow = (wave >> 2) << 4;   // 0 or 16
  const int wcol = (wave & 3) * 96;    // 0,96,192,288

  for (int i = tid; i < 32 * 128; i += 256) { h32[i] = 0.f; h16[i] = (_Float16)0.f; }
  float hsum[16];
#pragma unroll
  for (int k = 0; k < 16; ++k) hsum[k] = 0.f;
  __syncthreads();

  for (int t = 0; t < 32; ++t) {
    int tt = dir ? (31 - t) : t;
    // A rows for x_t: sequence r -> X row r*32 + tt  (lda = 32*128)
    const _Float16* Xbase = X + ((size_t)(rowbase + wrow) * 32 + tt) * 128;

    v16h ax[4], ah[4];
#pragma unroll
    for (int k = 0; k < 4; ++k) {
      ax[k] = load_a_frag(Xbase + k * 32, 32 * 128);
      ah[k] = load_a_frag(&h16[wrow * 128] + k * 32, 128);
    }
#pragma unroll
    for (int nt = 0; nt < 6; ++nt) {
      int col = wcol + nt * 16;
      v8f cx = {}, ch = {};
#pragma unroll
      for (int k = 0; k < 4; ++k) {
        v16h bx = load_b_frag(Wi + (size_t)col * 128 + k * 32, 128);
        cx = wmma_f16(ax[k], bx, cx);
        v16h bhf = load_b_frag(Wh + (size_t)col * 128 + k * 32, 128);
        ch = wmma_f16(ah[k], bhf, ch);
      }
      float bix = bi[col + (lane & 15)];
      float bhx = bh[col + (lane & 15)];
      store_c_bias(&xg[wrow * 384], 384, col, bix, cx);
      store_c_bias(&hg[wrow * 384], 384, col, bhx, ch);
    }
    __syncthreads();

    // GRU elementwise update: 32*128 elems, 16 per thread (fixed mapping)
#pragma unroll
    for (int k = 0; k < 16; ++k) {
      int idx = tid + k * 256;
      int r = idx >> 7, j = idx & 127;
      float xr = xg[r * 384 + j],       hr = hg[r * 384 + j];
      float xz = xg[r * 384 + 128 + j], hz = hg[r * 384 + 128 + j];
      float xn = xg[r * 384 + 256 + j], hn = hg[r * 384 + 256 + j];
      float rg = 1.f / (1.f + __expf(-(xr + hr)));
      float zg = 1.f / (1.f + __expf(-(xz + hz)));
      float ng = tanhf(xn + rg * hn);
      float hnew = (1.f - zg) * ng + zg * h32[idx];
      h32[idx] = hnew;
      h16[idx] = (_Float16)hnew;
      hsum[k] += hnew;
    }
    __syncthreads();
  }

  // text_features mean: fwd -> cols [0,128), bwd -> cols [128,256)
#pragma unroll
  for (int k = 0; k < 16; ++k) {
    int idx = tid + k * 256;
    int r = idx >> 7, j = idx & 127;
    float m = hsum[k] * (1.f / 32.f);
    size_t o = (size_t)(rowbase + r) * 256 + dir * 128 + j;
    tf[o]   = m;
    tf16[o] = (_Float16)m;
  }
}

// ---------------------------------------------------------------------------
// Sinkhorn: one block per batch. S = En*Vn^T precomputed (global).
// Km = exp(-(1-S)/reg) held in LDS (128x256 f32 = 128KB of the 320KB WGP LDS).
// ---------------------------------------------------------------------------
__global__ __launch_bounds__(256) void k_sinkhorn(const float* S, float* Tout,
                                                  float* loss) {
  const int K = 128, M = 256;
  __shared__ float Km[128 * 256];
  __shared__ float u[128];
  __shared__ float v[256];
  __shared__ float wsum[8];
  int b = blockIdx.x;
  int tid = threadIdx.x, lane = tid & 31;
  const float* Sb = S + (size_t)b * K * M;

  for (int i = tid; i < K * M; i += 256) Km[i] = __expf(-(1.f - Sb[i]) * 20.f); // 1/0.05
  if (tid < M) v[tid] = 1.f / M;
  __syncthreads();

  for (int it = 0; it < 10; ++it) {
    if (tid < K) {
      float s = 0.f;
      for (int m = 0; m < M; ++m) s += Km[tid * M + m] * v[m];
      u[tid] = (1.f / K) / (s + 1e-8f);
    }
    __syncthreads();
    {
      float s = 0.f;
      for (int k = 0; k < K; ++k) s += Km[k * M + tid] * u[k];
      v[tid] = (1.f / M) / (s + 1e-8f);
    }
    __syncthreads();
  }

  float part = 0.f;
  for (int i = tid; i < K * M; i += 256) {
    int k = i >> 8, m = i & 255;
    float T = u[k] * Km[i] * v[m];
    Tout[(size_t)b * K * M + i] = T;
    part += T * (1.f - Sb[i]);
  }
#pragma unroll
  for (int off = 16; off; off >>= 1) part += __shfl_xor(part, off, 32);
  if (lane == 0) wsum[tid >> 5] = part;
  __syncthreads();
  if (tid == 0) {
    float t = 0.f;
    for (int i = 0; i < 8; ++i) t += wsum[i];
    atomicAdd(loss, t * (1.f / 16.f));
  }
}

// ---------------------------------------------------------------------------
// Host orchestration
// ---------------------------------------------------------------------------
extern "C" void kernel_launch(void* const* d_in, const int* in_sizes, int n_in,
                              void* d_out, int out_size, void* d_ws,
                              size_t ws_size, hipStream_t stream) {
  (void)in_sizes; (void)n_in; (void)out_size; (void)ws_size;

  const int*   ids    = (const int*)d_in[0];
  const float* video  = (const float*)d_in[2];   // (16,256,1024)
  const float* emb    = (const float*)d_in[3];
  const float* WiF    = (const float*)d_in[4];
  const float* WhF    = (const float*)d_in[5];
  const float* biF    = (const float*)d_in[6];
  const float* bhF    = (const float*)d_in[7];
  const float* WiB    = (const float*)d_in[8];
  const float* WhB    = (const float*)d_in[9];
  const float* biB    = (const float*)d_in[10];
  const float* bhB    = (const float*)d_in[11];
  const float* tprojW = (const float*)d_in[12];
  const float* tprojB = (const float*)d_in[13];
  const float* vprojW = (const float*)d_in[14];
  const float* vprojB = (const float*)d_in[15];
  const float* tsumW1 = (const float*)d_in[16];
  const float* tsumB1 = (const float*)d_in[17];
  const float* tsumW2 = (const float*)d_in[18];
  const float* tsumB2 = (const float*)d_in[19];
  const float* vsumW1 = (const float*)d_in[20];
  const float* vsumB1 = (const float*)d_in[21];
  const float* vsumW2 = (const float*)d_in[22];
  const float* vsumB2 = (const float*)d_in[23];
  const float* vtsW1  = (const float*)d_in[24];
  const float* vtsB1  = (const float*)d_in[25];
  const float* vtsW2  = (const float*)d_in[26];
  const float* vtsB2  = (const float*)d_in[27];

  float* out  = (float*)d_out;
  float* tseg = out;                 // (16,128)  zeros
  float* tf   = out + 2048;          // (16,128,256)
  float* vseg = out + 526336;        // (16,256)
  float* tsum = out + 530432;        // (16,128)
  float* vsum = out + 532480;        // (16,256)
  float* loss = out + 536576;        // scalar
  float* Tmat = out + 536577;        // (16,128,256)

  char* w = (char*)d_ws;
  auto alloc = [&](size_t bytes) {
    char* p = w;
    w += (bytes + 255) & ~(size_t)255;
    return p;
  };
  _Float16* wiF16   = (_Float16*)alloc(49152 * 2);
  _Float16* whF16   = (_Float16*)alloc(49152 * 2);
  _Float16* wiB16   = (_Float16*)alloc(49152 * 2);
  _Float16* whB16   = (_Float16*)alloc(49152 * 2);
  _Float16* vtsW16  = (_Float16*)alloc(262144 * 2);
  _Float16* vsumW16 = (_Float16*)alloc(262144 * 2);
  _Float16* tsumW16 = (_Float16*)alloc(32768 * 2);
  _Float16* tprW16  = (_Float16*)alloc(65536 * 2);
  _Float16* vprW16  = (_Float16*)alloc(262144 * 2);
  _Float16* X16     = (_Float16*)alloc((size_t)8388608 * 2);
  _Float16* tf16    = (_Float16*)alloc(524288 * 2);
  _Float16* V16     = (_Float16*)alloc((size_t)4194304 * 2);
  float*    hidVts  = (float*)alloc((size_t)1048576 * 4);
  float*    hidVsum = (float*)alloc((size_t)1048576 * 4);
  float*    hidTsum = (float*)alloc((size_t)262144 * 4);
  float*    pTsum   = (float*)alloc(2048 * 4);
  float*    pVsum   = (float*)alloc(4096 * 4);
  _Float16* Ew16    = (_Float16*)alloc(524288 * 2);
  _Float16* Vw16    = (_Float16*)alloc((size_t)4194304 * 2);
  float*    Eproj   = (float*)alloc((size_t)524288 * 4);
  float*    Vproj   = (float*)alloc((size_t)1048576 * 4);
  _Float16* En16    = (_Float16*)alloc(524288 * 2);
  _Float16* Vn16    = (_Float16*)alloc((size_t)1048576 * 2);
  float*    Smat    = (float*)alloc((size_t)524288 * 4);

  auto cvt = [&](const float* s, _Float16* d, long n) {
    k_f32_to_f16<<<dim3((unsigned)((n + 255) / 256)), dim3(256), 0, stream>>>(s, d, n);
  };

  // 1) weight + video f16 staging
  cvt(WiF, wiF16, 49152);   cvt(WhF, whF16, 49152);
  cvt(WiB, wiB16, 49152);   cvt(WhB, whB16, 49152);
  cvt(vtsW1, vtsW16, 262144);
  cvt(vsumW1, vsumW16, 262144);
  cvt(tsumW1, tsumW16, 32768);
  cvt(tprojW, tprW16, 65536);
  cvt(vprojW, vprW16, 262144);
  cvt(video, V16, 4194304);

  // 2) embedding gather
  k_gather<<<dim3(32768), dim3(256), 0, stream>>>(ids, emb, X16, (long)8388608);

  // 3) zero-fill outputs that are pure zeros / accumulated
  k_zero_f32<<<dim3(8), dim3(256), 0, stream>>>(tseg, 2048);
  k_zero_f32<<<dim3(1), dim3(256), 0, stream>>>(loss, 1);

  // 4) persistent bidirectional GRU scan (WMMA inside)
  k_gru_scan<<<dim3(128), dim3(256), 0, stream>>>(
      X16, wiF16, whF16, biF, bhF, wiB16, whB16, biB, bhB, tf, tf16);

  auto gemm = [&](const _Float16* A, const _Float16* W_, const float* bias,
                  float* D, int M, int N, int K, int batches, size_t sA,
                  size_t sW, size_t sD) {
    int tiles = (M >> 5) * (N >> 6);   // 32x64 tile per wave
    dim3 grid((unsigned)((tiles + 7) / 8), (unsigned)batches);
    k_wmma_gemm<<<grid, dim3(256), 0, stream>>>(A, W_, bias, D, M, N, K, sA, sW, sD);
  };

  // 5) scorer hidden layers (WMMA)
  gemm(V16, vtsW16, vtsB1, hidVts, 4096, 256, 1024, 1, 0, 0, 0);
  gemm(V16, vsumW16, vsumB1, hidVsum, 4096, 256, 1024, 1, 0, 0, 0);
  gemm(tf16, tsumW16, tsumB1, hidTsum, 2048, 128, 256, 1, 0, 0, 0);

  // 6) scorer reductions (act: 0=relu, 1=tanh)
  k_scorer_reduce<<<dim3(512), dim3(256), 0, stream>>>(hidVts, vtsW2, vtsB2,
                                                       vseg, (float*)nullptr,
                                                       4096, 256, 0);
  k_scorer_reduce<<<dim3(512), dim3(256), 0, stream>>>(hidVsum, vsumW2, vsumB2,
                                                       vsum, pVsum, 4096, 256, 1);
  k_scorer_reduce<<<dim3(256), dim3(256), 0, stream>>>(hidTsum, tsumW2, tsumB2,
                                                       tsum, pTsum, 2048, 128, 1);

  // 7) summary-weighted features -> f16
  k_scale_rows<<<dim3(2048), dim3(256), 0, stream>>>(tf, pTsum, Ew16, 2048, 256);
  k_scale_rows<<<dim3(16384), dim3(256), 0, stream>>>(video, pVsum, Vw16, 4096, 1024);

  // 8) alignment projections (WMMA)
  gemm(Ew16, tprW16, tprojB, Eproj, 2048, 256, 256, 1, 0, 0, 0);
  gemm(Vw16, vprW16, vprojB, Vproj, 4096, 256, 1024, 1, 0, 0, 0);

  // 9) row-normalize for cosine cost
  k_rownorm<<<dim3(256), dim3(256), 0, stream>>>(Eproj, En16, 2048, 256);
  k_rownorm<<<dim3(512), dim3(256), 0, stream>>>(Vproj, Vn16, 4096, 256);

  // 10) batched similarity S[b] = En[b] (128x256) * Vn[b]^T (WMMA, batch=16)
  gemm(En16, Vn16, nullptr, Smat, 128, 256, 256, 16,
       (size_t)128 * 256, (size_t)256 * 256, (size_t)128 * 256);

  // 11) Sinkhorn iterations + transport plan + OT loss
  k_sinkhorn<<<dim3(16), dim3(256), 0, stream>>>(Smat, Tmat, loss);
}
